// MultiHeadAttention_16621523435614
// MI455X (gfx1250) — matrix operand order repaired and checked by reading, not compiled, not run on hardware
//
#include <hip/hip_runtime.h>
#include <type_traits>

// Fixed shapes from the reference.
constexpr int Bc  = 4;
constexpr int Sc  = 2048;
constexpr int Dc  = 1024;
constexpr int Hc  = 16;
constexpr int HDc = 64;
constexpr int Mrows = Bc * Sc;   // 8192
constexpr int NTc = Dc / 16;     // 64 n-tiles per weight k-block

typedef __attribute__((ext_vector_type(16))) _Float16 v16h;
typedef __attribute__((ext_vector_type(8)))  _Float16 v8h;
typedef __attribute__((ext_vector_type(8)))  float    v8f;

__device__ __forceinline__ v16h join8(v8h lo, v8h hi) {
  return __builtin_shufflevector(lo, hi, 0, 1, 2, 3, 4, 5, 6, 7,
                                         8, 9, 10, 11, 12, 13, 14, 15);
}

__device__ __forceinline__ v8f wmma_f16(v16h a, v16h b, v8f c) {
  return __builtin_amdgcn_wmma_f32_16x16x32_f16(false, a, false, b,
                                                (short)0, c, false, false);
}

// ---- DPP16 16-lane reductions (stay within each 16-lane half of the wave) --
template <int CTRL>
__device__ __forceinline__ float dpp_mov_f32(float x) {
  return __int_as_float(__builtin_amdgcn_update_dpp(
      0, __float_as_int(x), CTRL, 0xf, 0xf, true));
}
__device__ __forceinline__ float rowmax16(float x) {
  x = fmaxf(x, dpp_mov_f32<0xB1>(x));    // quad_perm(1,0,3,2)
  x = fmaxf(x, dpp_mov_f32<0x4E>(x));    // quad_perm(2,3,0,1)
  x = fmaxf(x, dpp_mov_f32<0x141>(x));   // row_half_mirror
  x = fmaxf(x, dpp_mov_f32<0x140>(x));   // row_mirror
  return x;
}
__device__ __forceinline__ float rowsum16(float x) {
  x += dpp_mov_f32<0xB1>(x);
  x += dpp_mov_f32<0x4E>(x);
  x += dpp_mov_f32<0x141>(x);
  x += dpp_mov_f32<0x140>(x);
  return x;
}

// ---------------------------------------------------------------------------
// WMMA wave32 fragment layouts used here (A/B k order contiguous, not the ISA table order):
// A (16x32): lane<16 -> row=lane,     K = 0..15 (linear)
//            lane>=16 -> row=lane-16, K = 16..31
// B (32x16): lane<16 -> col=lane,     K rows 0..15 (linear)
//            lane>=16 -> col=lane-16, K rows 16..31
// C  (16x16): vgpr r: lanes 0..15 -> (M=r, N=lane); lanes16..31 -> (M=r+8)
// ---------------------------------------------------------------------------

__device__ __forceinline__ v16h load_a_frag_f16(const _Float16* __restrict__ rowptr,
                                                int k0, int lane) {
  const int off = (lane >= 16) ? 16 : 0;
  const _Float16* p = rowptr + k0 + off;
  v8h lo = *(const v8h*)(p);
  v8h hi = *(const v8h*)(p + 8);
  return join8(lo, hi);
}

// ---------------------------------------------------------------------------
// One-time f32 -> f16 conversion (row-major preserved). 4 elems/thread.
// ---------------------------------------------------------------------------
__global__ __launch_bounds__(256)
void cvt_f32_f16_kernel(const float* __restrict__ in, _Float16* __restrict__ out) {
  const int i = (blockIdx.x * 256 + threadIdx.x) * 4;
#pragma unroll
  for (int j = 0; j < 4; ++j) out[i + j] = (_Float16)in[i + j];
}

// ---------------------------------------------------------------------------
// Weight pre-pack: W f32 [K=1024, N=1024] row-major -> fragment-major f16,
// times `scale` (folds 1/sqrt(HD) into Wq). packed[((kt*NT+nt)*32+lane)*16+j].
// ---------------------------------------------------------------------------
__global__ __launch_bounds__(256)
void pack_w_kernel(const float* __restrict__ W, _Float16* __restrict__ Wp,
                   float scale) {
  const int i = blockIdx.x * 256 + threadIdx.x;   // 1M elements total
  const int j    = i & 15;
  const int lane = (i >> 4) & 31;
  const int tile = i >> 9;
  const int nt = tile & (NTc - 1);
  const int kt = tile >> 6;                        // / NTc
  const int col  = nt * 16 + (lane & 15);
  const int krow = kt * 32 + ((lane >= 16) ? 16 : 0) + j;
  Wp[i] = (_Float16)(W[(size_t)krow * Dc + col] * scale);
}

// ---------------------------------------------------------------------------
// GEMM: out[M=8192, N=1024] = A_f16[M,1024] @ Wp + bias*bscale
// Each wave computes 64x64: 4 A-frags x 4 B-frags -> 16 WMMAs per 16 b128
// loads per K-step; acc = 128 VGPRs (stays within the 256-reg window).
// MODE 0: f32 row-major [M,N]        (final projection -> d_out)
// MODE 1: f16 head-split [BH,S,HD]   (Q, K)
// MODE 2: f16 transposed [BH,HD,S]   (V, key-contiguous for P@V fragments)
// ---------------------------------------------------------------------------
template <int MODE>
__global__ __launch_bounds__(128)
void gemm64x64_kernel(const _Float16* __restrict__ A, const _Float16* __restrict__ Wp,
                      const float* __restrict__ bias, float bscale,
                      void* __restrict__ out) {
  const int lane = threadIdx.x & 31;
  const int wave = threadIdx.x >> 5;
  const int m0 = blockIdx.x * 64;
  const int n0 = blockIdx.y * 256 + wave * 64;
  const int ntile0 = n0 >> 4;

  const _Float16* arow = A + (size_t)(m0 + (lane & 15)) * Dc;

  v8f acc[4][4] = {};

  for (int kk = 0; kk < Dc; kk += 32) {
    const int kt = kk >> 5;
    if (kk + 32 < Dc) __builtin_prefetch(arow + kk + 32, 0, 0);
    v16h a[4];
#pragma unroll
    for (int i = 0; i < 4; ++i)
      a[i] = load_a_frag_f16(arow + (size_t)(i * 16) * Dc, kk, lane);
    const _Float16* bp = Wp + (((size_t)kt * NTc + ntile0) * 32 + lane) * 16;
#pragma unroll
    for (int t = 0; t < 4; ++t) {
      v16h b = *(const v16h*)(bp + (size_t)t * 512);   // 32 lanes * 16 halves
#pragma unroll
      for (int i = 0; i < 4; ++i)
        acc[i][t] = wmma_f16(a[i], b, acc[i][t]);
    }
  }

  const int rbase = (lane >= 16) ? 8 : 0;
#pragma unroll
  for (int mr = 0; mr < 4; ++mr) {
#pragma unroll
    for (int t = 0; t < 4; ++t) {
      const int col = n0 + t * 16 + (lane & 15);
      const float bv = bias[col] * bscale;
#pragma unroll
      for (int r = 0; r < 8; ++r) {
        const int m = m0 + mr * 16 + rbase + r;
        const float val = acc[mr][t][r] + bv;
        if (MODE == 0) {
          ((float*)out)[(size_t)m * Dc + col] = val;
        } else {
          const int bb = m >> 11;          // / Sc
          const int ss = m & (Sc - 1);
          const int hh = col >> 6;
          const int hd = col & 63;
          if (MODE == 1) {
            ((_Float16*)out)[(((size_t)(bb * Hc + hh)) * Sc + ss) * HDc + hd] =
                (_Float16)val;
          } else {
            ((_Float16*)out)[(((size_t)(bb * Hc + hh)) * HDc + hd) * Sc + ss] =
                (_Float16)val;
          }
        }
      }
    }
  }
}

// ---------------------------------------------------------------------------
// Causal flash attention. Q (pre-scaled by 1/8), K f16 [BH,S,HD]; V f16
// [BH,HD,S]. One wave per block, 16-row q tile, key tiles of 32, online
// softmax in f32 with DPP row reductions. The key stream is q0>>5 fully-
// unmasked tiles plus exactly one masked tail tile.
// ---------------------------------------------------------------------------
__global__ __launch_bounds__(32)
void flash_attn_kernel(const _Float16* __restrict__ Q, const _Float16* __restrict__ K,
                       const _Float16* __restrict__ V, _Float16* __restrict__ Z) {
  __shared__ _Float16 Pld[16][40];   // stride 40 halves keeps 16B alignment

  const int lane = threadIdx.x;
  const int half = (lane >= 16) ? 1 : 0;
  const int q0 = blockIdx.x * 16;
  const int bh = blockIdx.y;
  const int bb = bh >> 4;
  const int hh = bh & 15;

  const _Float16* qbase = Q + (size_t)bh * Sc * HDc;
  const _Float16* kbase = K + (size_t)bh * Sc * HDc;
  const _Float16* vbase = V + (size_t)bh * HDc * Sc;

  const _Float16* qrow = qbase + (size_t)(q0 + (lane & 15)) * HDc;
  v16h qa0 = load_a_frag_f16(qrow, 0, lane);
  v16h qa1 = load_a_frag_f16(qrow, 32, lane);

  v8f o[4] = {};
  float mI[8], lI[8];
#pragma unroll
  for (int r = 0; r < 8; ++r) { mI[r] = -3.0e38f; lI[r] = 0.0f; }

  auto tile = [&](int k0, auto masked_c) {
    constexpr bool MASKED = decltype(masked_c)::value;

    // ---- scores: two 16x16 tiles, K^T fragments contiguous along d ----
    v8f s[2];
#pragma unroll
    for (int t = 0; t < 2; ++t) {
      const int key = k0 + t * 16 + (lane & 15);
      const _Float16* kp = kbase + (size_t)key * HDc + half * 16;
      v16h kt0 = *(const v16h*)(kp);        // d = half*16 + 0..15
      v16h kt1 = *(const v16h*)(kp + 32);   // d = 32 + half*16 + 0..15
      v8f z = {};
      z = wmma_f16(qa0, kt0, z);
      z = wmma_f16(qa1, kt1, z);
      s[t] = z;
    }

    // ---- causal mask only on the diagonal tile (scale folded into Wq) ----
    if (MASKED) {
#pragma unroll
      for (int t = 0; t < 2; ++t) {
        const int kcol = k0 + t * 16 + (lane & 15);
#pragma unroll
        for (int r = 0; r < 8; ++r) {
          const int qr = q0 + half * 8 + r;
          s[t][r] = (kcol <= qr) ? s[t][r] : -3.0e38f;
        }
      }
    }

    // ---- online softmax: DPP row reductions across 16-lane halves ----
    float corr[8];
#pragma unroll
    for (int r = 0; r < 8; ++r) {
      const float rm = rowmax16(fmaxf(s[0][r], s[1][r]));
      const float mn = fmaxf(mI[r], rm);
      corr[r] = __expf(mI[r] - mn);
      mI[r] = mn;
      const float p0 = __expf(s[0][r] - mn);
      const float p1 = __expf(s[1][r] - mn);
      s[0][r] = p0;
      s[1][r] = p1;
      lI[r] = lI[r] * corr[r] + rowsum16(p0 + p1);
    }

    // ---- transpose P (C layout) -> A layout via f16 LDS ----
    __syncthreads();
#pragma unroll
    for (int t = 0; t < 2; ++t)
#pragma unroll
      for (int r = 0; r < 8; ++r)
        Pld[half * 8 + r][t * 16 + (lane & 15)] = (_Float16)s[t][r];
    __syncthreads();

    v16h pa;
    {
      const int prow = lane & 15;
      const int koff = half ? 16 : 0;
      v8h lo = *(const v8h*)(&Pld[prow][koff]);
      v8h hi = *(const v8h*)(&Pld[prow][koff + 8]);
      pa = join8(lo, hi);
    }

    // ---- O = O*corr + P @ V; V fragments contiguous along keys ----
#pragma unroll
    for (int f = 0; f < 4; ++f) {
#pragma unroll
      for (int r = 0; r < 8; ++r) o[f][r] *= corr[r];
      const int dcol = f * 16 + (lane & 15);
      v16h vb = *(const v16h*)(vbase + (size_t)dcol * Sc + k0 + half * 16);
      o[f] = wmma_f16(pa, vb, o[f]);
    }
  };

  // Fully-unmasked tiles (k0 + 31 <= q0), then one masked diagonal tile.
  const int nFull = q0 >> 5;
  int k0 = 0;
  for (int i = 0; i < nFull; ++i, k0 += 32)
    tile(k0, std::false_type{});
  tile(k0, std::true_type{});

  // ---- normalize, write Z as f16 row-major [M, D] (merged heads) ----
#pragma unroll
  for (int f = 0; f < 4; ++f)
#pragma unroll
    for (int r = 0; r < 8; ++r) {
      const int qr = q0 + half * 8 + r;
      const int dm = hh * HDc + f * 16 + (lane & 15);
      Z[((size_t)(bb * Sc + qr)) * Dc + dm] = (_Float16)(o[f][r] / lI[r]);
    }
}

// ---------------------------------------------------------------------------
// Launch: cvt inputs -> pack weights -> 3 projections -> attention -> output.
// ---------------------------------------------------------------------------
extern "C" void kernel_launch(void* const* d_in, const int* in_sizes, int n_in,
                              void* d_out, int out_size, void* d_ws, size_t ws_size,
                              hipStream_t stream) {
  (void)in_sizes; (void)n_in; (void)out_size; (void)ws_size;
  const float* data    = (const float*)d_in[0];
  const float* context = (const float*)d_in[1];
  const float* Wq = (const float*)d_in[2];
  const float* bq = (const float*)d_in[3];
  const float* Wk = (const float*)d_in[4];
  const float* bk = (const float*)d_in[5];
  const float* Wv = (const float*)d_in[6];
  const float* bv = (const float*)d_in[7];
  const float* Wo = (const float*)d_in[8];
  const float* bo = (const float*)d_in[9];

  _Float16* ws = (_Float16*)d_ws;
  const size_t tsz = (size_t)Mrows * Dc;   // 8,388,608 elements
  const size_t wsz = (size_t)Dc * Dc;      // 1,048,576 elements
  _Float16* data16 = ws;
  _Float16* ctx16  = data16 + tsz;
  _Float16* Wqp    = ctx16 + tsz;
  _Float16* Wkp    = Wqp + wsz;
  _Float16* Wvp    = Wkp + wsz;
  _Float16* Wop    = Wvp + wsz;
  _Float16* Qh     = Wop + wsz;
  _Float16* Kh     = Qh + tsz;
  _Float16* Vt     = Kh + tsz;
  _Float16* Zb     = Vt + tsz;

  // 1) inputs to f16
  cvt_f32_f16_kernel<<<dim3(tsz / 1024), dim3(256), 0, stream>>>(data, data16);
  cvt_f32_f16_kernel<<<dim3(tsz / 1024), dim3(256), 0, stream>>>(context, ctx16);

  // 2) fragment-pack weights to f16 (Wq carries the 1/sqrt(64) scale)
  dim3 pg(wsz / 256);
  pack_w_kernel<<<pg, dim3(256), 0, stream>>>(Wq, Wqp, 0.125f);
  pack_w_kernel<<<pg, dim3(256), 0, stream>>>(Wk, Wkp, 1.0f);
  pack_w_kernel<<<pg, dim3(256), 0, stream>>>(Wv, Wvp, 1.0f);
  pack_w_kernel<<<pg, dim3(256), 0, stream>>>(Wo, Wop, 1.0f);

  // 3) projections (Q,K head-split; V transposed key-major)
  dim3 gblock(128);
  dim3 ggrid(Mrows / 64, Dc / 256);   // 128 x 4
  gemm64x64_kernel<1><<<ggrid, gblock, 0, stream>>>(data16, Wqp, bq, 0.125f, Qh);
  gemm64x64_kernel<1><<<ggrid, gblock, 0, stream>>>(ctx16,  Wkp, bk, 1.0f,  Kh);
  gemm64x64_kernel<2><<<ggrid, gblock, 0, stream>>>(ctx16,  Wvp, bv, 1.0f,  Vt);

  // 4) causal flash attention (16 queries per wave; DPP softmax reductions)
  dim3 agrid(Sc / 16, Bc * Hc);       // 128 x 64
  flash_attn_kernel<<<agrid, dim3(32), 0, stream>>>(Qh, Kh, Vt, Zb);

  // 5) output projection (f32 + bias)
  gemm64x64_kernel<0><<<ggrid, gblock, 0, stream>>>(Zb, Wop, bo, 1.0f, d_out);
}
